// LocalHilbertAggregator_87479893885013
// MI455X (gfx1250) — compile-verified, matrix operand order checked by reading, not run on hardware
//
#include <hip/hip_runtime.h>

// ---------------------------------------------------------------------------
// Problem constants (from reference)
// ---------------------------------------------------------------------------
#define BATCH   64
#define SEQL    1024
#define DIMM    256
#define DSTATE  16
#define DCONV   4
#define DINNER  512           // EXPAND * DIM
#define NHEADS  8
#define HEADD   64
#define DPROJ   1064          // 2*DINNER + 2*DSTATE + NHEADS
#define CCH     544           // DINNER + 2*DSTATE
#define WIDTH_  640
#define HEIGHT_ 480

static const size_t BL = (size_t)BATCH * SEQL;   // 65536 token rows

// ---------------------------------------------------------------------------
// bf16 helpers / WMMA types
// ---------------------------------------------------------------------------
typedef __attribute__((ext_vector_type(16))) __bf16 v16bf;
typedef __attribute__((ext_vector_type(8)))  __bf16 v8bf;
typedef __attribute__((ext_vector_type(8)))  float  v8f;

union BFrag { v16bf v; v8bf h[2]; };

__device__ __forceinline__ unsigned short f2bf(float f) {
    union { float f; unsigned int u; } c; c.f = f;
    unsigned int u = c.u;
    unsigned int r = (u + 0x7FFFu + ((u >> 16) & 1u)) >> 16;  // RNE
    return (unsigned short)r;
}

__device__ __forceinline__ float silu_f(float v) {
    return v / (1.0f + __expf(-v));
}

// ---------------------------------------------------------------------------
// K0: f32 -> bf16 weight conversion (one element per thread)
// ---------------------------------------------------------------------------
__global__ void convert_bf16_kernel(const float* __restrict__ src,
                                    unsigned short* __restrict__ dst, int n) {
    int i = blockIdx.x * 256 + threadIdx.x;
    if (i < n) dst[i] = f2bf(src[i]);
}

// ---------------------------------------------------------------------------
// K1: stable argsort per (batch, ordering) via bitonic sort of composite keys
// grid (BATCH, 2) x 256 threads
// ---------------------------------------------------------------------------
__global__ __launch_bounds__(256) void sort_kernel(const int* __restrict__ coords,
                                                   int* __restrict__ perm,
                                                   int* __restrict__ inv) {
    __shared__ unsigned int key[SEQL];
    const int b = blockIdx.x, ord = blockIdx.y, tid = threadIdx.x;
    for (int i = tid; i < SEQL; i += 256) {
        int cx = coords[((size_t)b * SEQL + i) * 2 + 0];
        int cy = coords[((size_t)b * SEQL + i) * 2 + 1];
        unsigned int k = (ord == 0) ? (unsigned int)(cy * WIDTH_ + cx)
                                    : (unsigned int)(cx * HEIGHT_ + cy);
        key[i] = (k << 10) | (unsigned int)i;   // stable: index as tiebreaker
    }
    __syncthreads();
    for (unsigned int k = 2; k <= SEQL; k <<= 1) {
        for (unsigned int j = k >> 1; j > 0; j >>= 1) {
            for (int i = tid; i < SEQL; i += 256) {
                unsigned int ixj = (unsigned int)i ^ j;
                if (ixj > (unsigned int)i) {
                    unsigned int a = key[i], c = key[ixj];
                    bool asc = ((i & k) == 0);
                    if ((a > c) == asc) { key[i] = c; key[ixj] = a; }
                }
            }
            __syncthreads();
        }
    }
    const int base = (b * 2 + ord) * SEQL;
    for (int i = tid; i < SEQL; i += 256) {
        int p = (int)(key[i] & 1023u);
        perm[base + i] = p;
        inv[base + p]  = i;
    }
}

// ---------------------------------------------------------------------------
// K2: gather permuted tokens, convert to bf16  (grid BL x 256)
// ---------------------------------------------------------------------------
__global__ __launch_bounds__(256) void gather_kernel(const float* __restrict__ x,
                                                     const int* __restrict__ perm,
                                                     unsigned short* __restrict__ u,
                                                     int ord) {
    const int row = blockIdx.x;            // b*SEQL + l
    const int b = row >> 10, l = row & 1023;
    const int src = perm[(b * 2 + ord) * SEQL + l];
    float v = x[((size_t)b * SEQL + src) * DIMM + threadIdx.x];
    u[(size_t)row * DIMM + threadIdx.x] = f2bf(v);
}

// ---------------------------------------------------------------------------
// K3: WMMA bf16 GEMM  C[M,N] = A[M,K] * W[N,K]^T   (f32 accumulate)
// Each wave owns one 16-row M tile and NTW consecutive 16-col N tiles, so a
// single A fragment register set is reused across NTW WMMAs per K step.
// 256 threads = 8 waves; block covers 8*NTW N tiles.
// grid: (M/16, ceil(Ntiles / (8*NTW)))
// ---------------------------------------------------------------------------
template<int NTW, bool NTSTORE>
__global__ __launch_bounds__(256) void wmma_gemm_bf16(
        const unsigned short* __restrict__ A,
        const unsigned short* __restrict__ W,
        float* __restrict__ C, int N, int K, int Ntiles) {
    const int wave   = threadIdx.x >> 5;
    const int lane   = threadIdx.x & 31;
    const int mtile  = blockIdx.x;
    const int ntbase = (blockIdx.y * 8 + wave) * NTW;
    if (ntbase >= Ntiles) return;           // wave-uniform cull (EXEC stays full)

    // ISA 7.12.2 16-bit A/B layout: lane holds M(or N)=lane&15,
    // K runs: [kgrp, kgrp+8) and [kgrp+16, kgrp+24), kgrp = (lane>>4)*8.
    const int mrow = (mtile << 4) + (lane & 15);
    const int kgrp = (lane >> 4) << 3;
    const unsigned short* pa = A + (size_t)mrow * K + kgrp;

    const unsigned short* pb[NTW];
    int ncol[NTW];
    #pragma unroll
    for (int j = 0; j < NTW; ++j) {
        ncol[j] = ((ntbase + j) << 4) + (lane & 15);
        int cc  = (ncol[j] < N) ? ncol[j] : (N - 1);    // clamp edge-tile loads
        pb[j]   = W + (size_t)cc * K + kgrp;
    }

    v8f zero = {};
    v8f acc[NTW];
    #pragma unroll
    for (int j = 0; j < NTW; ++j) acc[j] = zero;

    for (int k0 = 0; k0 < K; k0 += 32) {
        BFrag a;
        a.h[0] = *(const v8bf*)(pa + k0);
        a.h[1] = *(const v8bf*)(pa + k0 + 16);
        #pragma unroll
        for (int j = 0; j < NTW; ++j) {
            BFrag bm;
            bm.h[0] = *(const v8bf*)(pb[j] + k0);
            bm.h[1] = *(const v8bf*)(pb[j] + k0 + 16);
            acc[j] = __builtin_amdgcn_wmma_f32_16x16x32_bf16(
                         false, a.v, false, bm.v, (short)0, acc[j], false, false);
        }
    }
    // C/D layout: VGPR r holds M = r + 8*(lane>>4), N = lane&15
    const int rbase = (mtile << 4) + ((lane >> 4) << 3);
    #pragma unroll
    for (int j = 0; j < NTW; ++j) {
        if (ncol[j] < N) {
            #pragma unroll
            for (int r = 0; r < 8; ++r) {
                float* dst = &C[(size_t)(rbase + r) * N + ncol[j]];
                if (NTSTORE) __builtin_nontemporal_store(acc[j][r], dst);
                else         *dst = acc[j][r];
            }
        }
    }
}

// ---------------------------------------------------------------------------
// K4: causal depthwise conv(4) + SiLU over xBC channels  (grid BL x 544)
// ---------------------------------------------------------------------------
__global__ __launch_bounds__(CCH) void conv_silu_kernel(
        const float* __restrict__ zx, const float* __restrict__ cw,
        const float* __restrict__ cb, float* __restrict__ xbc) {
    const int row = blockIdx.x;
    const int c   = threadIdx.x;
    const int l   = row & 1023;
    float acc = cb[c];
    #pragma unroll
    for (int t = 0; t < DCONV; ++t) {
        int li = l - (DCONV - 1) + t;
        if (li >= 0)
            acc += cw[c * DCONV + t] * zx[(size_t)(row - (DCONV - 1) + t) * DPROJ + DINNER + c];
    }
    xbc[(size_t)row * CCH + c] = silu_f(acc);
}

// ---------------------------------------------------------------------------
// K5: dt = softplus(raw + dt_bias)   (BL*NHEADS elementwise)
// ---------------------------------------------------------------------------
__global__ __launch_bounds__(256) void dt_kernel(const float* __restrict__ zx,
                                                 const float* __restrict__ dt_bias,
                                                 float* __restrict__ dtb) {
    const int idx = blockIdx.x * 256 + threadIdx.x;   // < BL*NHEADS
    const int row = idx >> 3, h = idx & 7;
    float v = zx[(size_t)row * DPROJ + (DPROJ - NHEADS) + h] + dt_bias[h];
    dtb[idx] = (v > 20.0f) ? v : log1pf(__expf(v));
}

// ---------------------------------------------------------------------------
// K6: sequential SSM scan. grid = BATCH*NHEADS blocks, 64 threads (one per p).
// Each thread keeps its 16-wide state column in registers.
// ---------------------------------------------------------------------------
__global__ __launch_bounds__(HEADD) void scan_kernel(
        const float* __restrict__ xbc, const float* __restrict__ dtb,
        const float* __restrict__ A_log, const float* __restrict__ Dp,
        float* __restrict__ y) {
    const int b = blockIdx.x >> 3, h = blockIdx.x & 7, p = threadIdx.x;
    __shared__ float sB[DSTATE], sC[DSTATE], sdt[1];
    float hreg[DSTATE];
    #pragma unroll
    for (int n = 0; n < DSTATE; ++n) hreg[n] = 0.0f;
    const float Ah = -__expf(A_log[h]);
    const float Dh = Dp[h];
    for (int l = 0; l < SEQL; ++l) {
        const size_t base = (size_t)b * SEQL + l;
        if (p < DSTATE)            sB[p]            = xbc[base * CCH + DINNER + p];
        else if (p < 2 * DSTATE)   sC[p - DSTATE]   = xbc[base * CCH + DINNER + DSTATE + (p - DSTATE)];
        else if (p == 2 * DSTATE)  sdt[0]           = dtb[base * NHEADS + h];
        __syncthreads();
        const float xp  = xbc[base * CCH + h * HEADD + p];
        const float dtv = sdt[0];
        const float dA  = __expf(dtv * Ah);
        const float sc  = dtv * xp;
        float yp = 0.0f;
        #pragma unroll
        for (int n = 0; n < DSTATE; ++n) {
            hreg[n] = hreg[n] * dA + sc * sB[n];
            yp += hreg[n] * sC[n];
        }
        y[base * DINNER + h * HEADD + p] = yp + Dh * xp;
        __syncthreads();
    }
}

// ---------------------------------------------------------------------------
// K7: y * silu(z), RMSNorm over 512, emit bf16 for out_proj (grid BL x 256)
// ---------------------------------------------------------------------------
__global__ __launch_bounds__(256) void gate_norm_kernel(
        const float* __restrict__ y, const float* __restrict__ zx,
        const float* __restrict__ norm_w, unsigned short* __restrict__ yg) {
    const int row = blockIdx.x, t = threadIdx.x;
    float z0 = zx[(size_t)row * DPROJ + t];
    float z1 = zx[(size_t)row * DPROJ + 256 + t];
    float v0 = y[(size_t)row * DINNER + t]       * silu_f(z0);
    float v1 = y[(size_t)row * DINNER + 256 + t] * silu_f(z1);
    __shared__ float red[256];
    red[t] = v0 * v0 + v1 * v1;
    __syncthreads();
    for (int o = 128; o > 0; o >>= 1) { if (t < o) red[t] += red[t + o]; __syncthreads(); }
    const float rms = rsqrtf(red[0] / (float)DINNER + 1e-5f);
    yg[(size_t)row * DINNER + t]       = f2bf(v0 * rms * norm_w[t]);
    yg[(size_t)row * DINNER + 256 + t] = f2bf(v1 * rms * norm_w[256 + t]);
}

// ---------------------------------------------------------------------------
// K8: inverse-permute + average both orderings, mean+max pool over L,
//     LayerNorm over DIM.  grid BATCH x 256 (one thread per output channel)
// ---------------------------------------------------------------------------
__global__ __launch_bounds__(256) void combine_kernel(
        const float* __restrict__ o0, const float* __restrict__ o1,
        const int* __restrict__ inv, const float* __restrict__ ln_w,
        const float* __restrict__ ln_b, float* __restrict__ out) {
    const int b = blockIdx.x, d = threadIdx.x;
    const int* inv0 = inv + (b * 2 + 0) * SEQL;
    const int* inv1 = inv + (b * 2 + 1) * SEQL;
    float s = 0.0f, mx = -3.0e38f;
    for (int l = 0; l < SEQL; ++l) {
        float v = 0.5f * (o0[((size_t)b * SEQL + inv0[l]) * DIMM + d] +
                          o1[((size_t)b * SEQL + inv1[l]) * DIMM + d]);
        s += v;
        mx = fmaxf(mx, v);
    }
    const float seg = 0.5f * (s * (1.0f / (float)SEQL) + mx);
    __shared__ float red[256];
    red[d] = seg;
    __syncthreads();
    for (int o = 128; o > 0; o >>= 1) { if (d < o) red[d] += red[d + o]; __syncthreads(); }
    const float mu = red[0] * (1.0f / (float)DIMM);
    __syncthreads();
    red[d] = (seg - mu) * (seg - mu);
    __syncthreads();
    for (int o = 128; o > 0; o >>= 1) { if (d < o) red[d] += red[d + o]; __syncthreads(); }
    const float var = red[0] * (1.0f / (float)DIMM);
    out[(size_t)b * DIMM + d] = (seg - mu) * rsqrtf(var + 1e-5f) * ln_w[d] + ln_b[d];
}

// ---------------------------------------------------------------------------
// Host driver
// ---------------------------------------------------------------------------
extern "C" void kernel_launch(void* const* d_in, const int* in_sizes, int n_in,
                              void* d_out, int out_size, void* d_ws, size_t ws_size,
                              hipStream_t stream) {
    (void)in_sizes; (void)n_in; (void)out_size; (void)ws_size;

    const float* x          = (const float*)d_in[0];
    const int*   coords     = (const int*)  d_in[1];
    const float* in_proj_w  = (const float*)d_in[2];
    const float* conv_w     = (const float*)d_in[3];
    const float* conv_b     = (const float*)d_in[4];
    const float* dt_bias    = (const float*)d_in[5];
    const float* A_log      = (const float*)d_in[6];
    const float* Dp         = (const float*)d_in[7];
    const float* norm_w     = (const float*)d_in[8];
    const float* out_proj_w = (const float*)d_in[9];
    const float* ln_w       = (const float*)d_in[10];
    const float* ln_b       = (const float*)d_in[11];
    float* out = (float*)d_out;

    // workspace carve-up (256B aligned)
    char*  ws  = (char*)d_ws;
    size_t cur = 0;
    auto alloc = [&](size_t bytes) -> size_t {
        size_t o = cur;
        cur += (bytes + 255) & ~(size_t)255;
        return o;
    };
    int*            perm = (int*)           (ws + alloc(2 * BATCH * SEQL * sizeof(int)));
    int*            inv  = (int*)           (ws + alloc(2 * BATCH * SEQL * sizeof(int)));
    unsigned short* w1b  = (unsigned short*)(ws + alloc((size_t)DPROJ * DIMM * 2));
    unsigned short* w2b  = (unsigned short*)(ws + alloc((size_t)DIMM * DINNER * 2));
    unsigned short* ubuf = (unsigned short*)(ws + alloc(BL * DIMM * 2));
    float*          zx   = (float*)         (ws + alloc(BL * DPROJ * 4));
    float*          xbc  = (float*)         (ws + alloc(BL * CCH * 4));
    float*          dtb  = (float*)         (ws + alloc(BL * NHEADS * 4));
    float*          ybuf = (float*)         (ws + alloc(BL * DINNER * 4));
    unsigned short* yg   = (unsigned short*)(ws + alloc(BL * DINNER * 2));
    float*          o0   = (float*)         (ws + alloc(BL * DIMM * 4));
    float*          o1   = (float*)         (ws + alloc(BL * DIMM * 4));

    // --- weight conversion (once per call, deterministic) ---
    {
        int n1 = DPROJ * DIMM;
        convert_bf16_kernel<<<(n1 + 255) / 256, 256, 0, stream>>>(in_proj_w, w1b, n1);
        int n2 = DIMM * DINNER;
        convert_bf16_kernel<<<(n2 + 255) / 256, 256, 0, stream>>>(out_proj_w, w2b, n2);
    }

    // --- stable argsort for both orderings ---
    sort_kernel<<<dim3(BATCH, 2), 256, 0, stream>>>(coords, perm, inv);

    const int MT  = (int)(BL / 16);        // 4096 M tiles
    const int NT1 = (DPROJ + 15) / 16;     // 67  N tiles (in_proj)
    const int NT2 = DIMM / 16;             // 16  N tiles (out_proj)

    for (int ord = 0; ord < 2; ++ord) {
        float* oo = (ord == 0) ? o0 : o1;

        gather_kernel<<<(int)BL, 256, 0, stream>>>(x, perm, ubuf, ord);

        // in_proj: 4 N-tiles per wave (A-fragment reuse), NT stores (C > L2)
        wmma_gemm_bf16<4, true><<<dim3(MT, (NT1 + 31) / 32), 256, 0, stream>>>(
            ubuf, w1b, zx, DPROJ, DIMM, NT1);

        conv_silu_kernel<<<(int)BL, CCH, 0, stream>>>(zx, conv_w, conv_b, xbc);

        dt_kernel<<<(int)(BL * NHEADS / 256), 256, 0, stream>>>(zx, dt_bias, dtb);

        scan_kernel<<<BATCH * NHEADS, HEADD, 0, stream>>>(xbc, dtb, A_log, Dp, ybuf);

        gate_norm_kernel<<<(int)BL, 256, 0, stream>>>(ybuf, zx, norm_w, yg);

        // out_proj: 2 N-tiles per wave => 8 waves cover all 16 N tiles exactly
        wmma_gemm_bf16<2, false><<<dim3(MT, (NT2 + 15) / 16), 256, 0, stream>>>(
            yg, w2b, oo, DIMM, DINNER, NT2);
    }

    combine_kernel<<<BATCH, 256, 0, stream>>>(o0, o1, inv, ln_w, ln_b, out);
}
